// FilterNormalizationConv2DParallel_43671227466090
// MI455X (gfx1250) — matrix operand order repaired and checked by reading, not compiled, not run on hardware
//
#include <hip/hip_runtime.h>
#include <hip/hip_bf16.h>

typedef __attribute__((ext_vector_type(16))) _Float16 v16h;
typedef __attribute__((ext_vector_type(8)))  _Float16 v8h;
typedef __attribute__((ext_vector_type(8)))  float    v8f;
typedef int v4i __attribute__((vector_size(16)));    // GCC-style, matches builtin's V4i

#define B_   8
#define C_   128
#define HW   112
#define L_   12544   /* 112*112 = 98*128 = 392*32 */
#define CO   256
#define KD   1152    /* 9*128 (weights reordered: pos-major, channel-minor) */
#define MT   128
#define NT   128
#define KBIG 128     /* K-block = all channels of one (kh,kw) position */
#define LDP  136     /* padded LDS row stride in f16 (272B: conflict-free b128 frag loads) */

#if defined(__gfx1250__) && __has_builtin(__builtin_amdgcn_global_load_async_to_lds_b128)
#define USE_ASYNC_LDS 1
#define AS1_V4I(p) ((__attribute__((address_space(1))) v4i*)(p))
#define AS3_V4I(p) ((__attribute__((address_space(3))) v4i*)(p))
#else
#define USE_ASYNC_LDS 0
#endif

#if defined(__gfx1250__) && __has_builtin(__builtin_amdgcn_s_wait_asynccnt)
#define ASYNC_WAIT0() __builtin_amdgcn_s_wait_asynccnt(0)
#elif defined(__gfx1250__)
#define ASYNC_WAIT0() asm volatile("s_wait_asynccnt 0" ::: "memory")
#else
#define ASYNC_WAIT0()
#endif

// ---------------- phase 0a: weight fp32 -> f16, reorder (c,kh,kw)->(kh*3+kw,c) ----------------
__global__ void cvt_weight_kernel(const float* __restrict__ w, _Float16* __restrict__ wf)
{
    int i = blockIdx.x * 256 + threadIdx.x;          // dst index: o*1152 + pos*128 + c
    if (i >= CO * KD) return;
    int o = i / KD, rem = i - o * KD;
    int pos = rem >> 7, c = rem & 127;
    wf[i] = (_Float16)w[(size_t)o * KD + c * 9 + pos];
}

// ---------------- phase 0b: x (B,C,L) f32 -> x16t (B,L,C) f16 (tiled transpose) ----------------
__global__ __launch_bounds__(256) void transpose_cvt_kernel(
    const float* __restrict__ x, _Float16* __restrict__ xt)
{
    __shared__ _Float16 t[32][33];                   // +1 pad: conflict-free transpose
    const int b  = blockIdx.z;
    const int c0 = blockIdx.y * 32;
    const int l0 = blockIdx.x * 32;
    const int tx = threadIdx.x & 31;
    const int ty = threadIdx.x >> 5;                 // 8 rows per pass
    const float* xb = x + (size_t)b * C_ * L_;
    #pragma unroll
    for (int r = 0; r < 32; r += 8)                  // read coalesced along l
        t[ty + r][tx] = (_Float16)xb[(size_t)(c0 + ty + r) * L_ + l0 + tx];
    __syncthreads();
    _Float16* xo = xt + (size_t)b * L_ * C_;
    #pragma unroll
    for (int r = 0; r < 32; r += 8)                  // write coalesced along c
        xo[(size_t)(l0 + ty + r) * C_ + c0 + tx] = t[tx][ty + r];
}

// ---------------- phase 1a: per-pixel channel sum of squares ----------------
__global__ void chansq_kernel(const float* __restrict__ x, float* __restrict__ ssum)
{
    int i = blockIdx.x * 256 + threadIdx.x;          // b*L + pix
    if (i >= B_ * L_) return;
    int b = i / L_, pix = i - b * L_;
    const float* p = x + (size_t)b * C_ * L_ + pix;
    float s = 0.f;
    #pragma unroll 4
    for (int c = 0; c < C_; ++c) { float v = p[(size_t)c * L_]; s += v * v; }
    ssum[i] = s;
}

// ---------------- phase 1b: 3x3 window sum -> 1/max(sqrt, eps) ----------------
__global__ void invnorm_kernel(const float* __restrict__ ssum, float* __restrict__ inv)
{
    int i = blockIdx.x * 256 + threadIdx.x;
    if (i >= B_ * L_) return;
    int b = i / L_, pix = i - b * L_;
    int oh = pix / HW, ow = pix - oh * HW;
    const float* s = ssum + (size_t)b * L_;
    float acc = 0.f;
    #pragma unroll
    for (int dh = -1; dh <= 1; ++dh)
        #pragma unroll
        for (int dw = -1; dw <= 1; ++dw) {
            int h = oh + dh, w = ow + dw;
            if ((unsigned)h < HW && (unsigned)w < HW) acc += s[h * HW + w];
        }
    inv[i] = 1.0f / fmaxf(sqrtf(acc), 1e-12f);
}

// ---------------- phase 1c: zero the 256B guard region (conv padding source) ----------------
__global__ void zero_guard_kernel(float* __restrict__ z)
{
    z[threadIdx.x] = 0.f;                            // 64 threads * 4B = 256B
}

// ---------------- phase 2: implicit-GEMM conv via f16 WMMA ----------------
__global__ __launch_bounds__(256) void conv_wmma_kernel(
    const _Float16* __restrict__ x16t, const _Float16* __restrict__ wf,
    const float* __restrict__ inv, const float* __restrict__ bias,
    const _Float16* __restrict__ zg, float* __restrict__ out)
{
#if USE_ASYNC_LDS
    __shared__ __align__(16) _Float16 As[2][MT * LDP];   // double-buffered (async engine)
    __shared__ __align__(16) _Float16 Bs[2][NT * LDP];
#else
    __shared__ __align__(16) _Float16 As[1][MT * LDP];   // single buffer (reg pipeline)
    __shared__ __align__(16) _Float16 Bs[1][NT * LDP];
#endif

    const int b    = blockIdx.z;
    const int mwg  = blockIdx.y * MT;
    const int nwg  = blockIdx.x * NT;
    const int tid  = threadIdx.x;
    const int lane = tid & 31;            // wave32
    const int wid  = tid >> 5;            // 8 waves
    const int wm   = (wid >> 2) * 64;     // wave M offset: 0 / 64
    const int wn   = (wid & 3)  * 32;     // wave N offset: 0/32/64/96
    const int lrow  = lane & 15;
    const int khalf = lane >> 4;

    v8f acc[4][2] = {};                   // 4 M-frags x 2 N-frags, f32 accum

    // ---- staging slot invariants: thread stages rows r = (tid>>4)+16*it, chunk ch = tid&15 ----
    const int rb = tid >> 4;              // base row 0..15
    const int ch = tid & 15;              // 16B chunk within 128-f16 row
    int poh[8], pow_[8];
    #pragma unroll
    for (int it = 0; it < 8; ++it) {
        int l = nwg + rb + it * 16;
        poh[it] = l / HW;
        pow_[it] = l - poh[it] * HW;
    }
    const _Float16* xb  = x16t + (size_t)b * L_ * C_;
    const _Float16* wfa = wf + (size_t)mwg * KD;

    // ---- 32 WMMAs from one staged 128x128 K-block ----
    auto compute = [&](const _Float16* Ab, const _Float16* Bb) {
        #pragma unroll
        for (int ks = 0; ks < 4; ++ks) {
            const int kofs = ks * 32;
            union { v16h v; v8h h[2]; } bf[2];
            #pragma unroll
            for (int ni = 0; ni < 2; ++ni) {
                int n = wn + ni * 16 + lrow;
                bf[ni].h[0] = *reinterpret_cast<const v8h*>(&Bb[n * LDP + kofs + khalf * 8]);
                bf[ni].h[1] = *reinterpret_cast<const v8h*>(&Bb[n * LDP + kofs + 16 + khalf * 8]);
            }
            #pragma unroll
            for (int mi = 0; mi < 4; ++mi) {
                int m = wm + mi * 16 + lrow;
                union { v16h v; v8h h[2]; } af;
                af.h[0] = *reinterpret_cast<const v8h*>(&Ab[m * LDP + kofs + khalf * 8]);
                af.h[1] = *reinterpret_cast<const v8h*>(&Ab[m * LDP + kofs + 16 + khalf * 8]);
                #pragma unroll
                for (int ni = 0; ni < 2; ++ni) {
                    acc[mi][ni] = __builtin_amdgcn_wmma_f32_16x16x32_f16(
                        false, af.v, false, bf[ni].v,
                        (short)0, acc[mi][ni], false, false);
                }
            }
        }
    };

#if USE_ASYNC_LDS
    // ===== async global->LDS double-buffered pipeline (ASYNCcnt) =====
    auto stage = [&](int pos, int buf) {
        const int kh = pos / 3, kw = pos - (pos / 3) * 3;
        const int koff = (kh - 1) * HW + (kw - 1);
        #pragma unroll
        for (int it = 0; it < 8; ++it) {
            int r = rb + it * 16;
            const _Float16* ga = wfa + (size_t)r * KD + pos * KBIG + ch * 8;
            __builtin_amdgcn_global_load_async_to_lds_b128(
                AS1_V4I(ga), AS3_V4I(&As[buf][r * LDP + ch * 8]), 0, 0);
            bool inb = ((unsigned)(poh[it] + kh - 1) < HW) &&
                       ((unsigned)(pow_[it] + kw - 1) < HW);
            const _Float16* gb = inb
                ? (xb + (size_t)(nwg + r + koff) * C_ + ch * 8)
                : (zg + ch * 8);                       // zero-guard: conv padding
            __builtin_amdgcn_global_load_async_to_lds_b128(
                AS1_V4I(gb), AS3_V4I(&Bs[buf][r * LDP + ch * 8]), 0, 0);
        }
    };

    stage(0, 0);
    #pragma unroll 1
    for (int pos = 0; pos < 9; ++pos) {
        const int buf = pos & 1;
        ASYNC_WAIT0();                    // own async copies for `buf` complete
        __syncthreads();                  // all waves' copies visible; prev compute done
        if (pos < 8) stage(pos + 1, buf ^ 1);   // fill other buffer behind the WMMAs
        compute(&As[buf][0], &Bs[buf][0]);
    }
#else
    // ===== fallback: register-pipelined staging =====
    uint4 pa[8], pb[8];
    const uint4 z4 = make_uint4(0u, 0u, 0u, 0u);
    #pragma unroll
    for (int it = 0; it < 8; ++it) {      // prefetch pos = 0 (kh=kw=0, koff=-113)
        int m = rb + it * 16;
        pa[it] = *reinterpret_cast<const uint4*>(wfa + (size_t)m * KD + ch * 8);
        bool inb = ((unsigned)(poh[it] - 1) < HW) && ((unsigned)(pow_[it] - 1) < HW);
        const uint4* s = reinterpret_cast<const uint4*>(
            xb + (size_t)(nwg + rb + it * 16 - 113) * C_) + ch;
        pb[it] = inb ? *s : z4;
    }
    #pragma unroll 1
    for (int pos = 0; pos < 9; ++pos) {
        #pragma unroll
        for (int it = 0; it < 8; ++it) {
            int r = rb + it * 16;
            *reinterpret_cast<uint4*>(&As[0][r * LDP + ch * 8]) = pa[it];
            *reinterpret_cast<uint4*>(&Bs[0][r * LDP + ch * 8]) = pb[it];
        }
        __syncthreads();
        if (pos < 8) {
            const int np = pos + 1;
            const int kh = np / 3, kw = np - (np / 3) * 3;
            const int koff = (kh - 1) * HW + (kw - 1);
            #pragma unroll
            for (int it = 0; it < 8; ++it) {
                int m = rb + it * 16;
                pa[it] = *reinterpret_cast<const uint4*>(
                    wfa + (size_t)m * KD + np * KBIG + ch * 8);
                bool inb = ((unsigned)(poh[it] + kh - 1) < HW) &&
                           ((unsigned)(pow_[it] + kw - 1) < HW);
                const uint4* s = reinterpret_cast<const uint4*>(
                    xb + (size_t)(nwg + rb + it * 16 + koff) * C_) + ch;
                pb[it] = inb ? *s : z4;
            }
        }
        compute(&As[0][0], &Bs[0][0]);
        __syncthreads();
    }
#endif

    // ---- epilogue: out = acc * invnorm[l] + bias[o] (non-temporal stream) ----
    #pragma unroll
    for (int ni = 0; ni < 2; ++ni) {
        int gn = nwg + wn + ni * 16 + lrow;                    // output column l
        float iv = inv[b * L_ + gn];
        #pragma unroll
        for (int mi = 0; mi < 4; ++mi) {
            int gm0 = mwg + wm + mi * 16 + khalf * 8;          // VGPR i -> M = gm0 + i
            #pragma unroll
            for (int i = 0; i < 8; ++i) {
                int o = gm0 + i;
                __builtin_nontemporal_store(acc[mi][ni][i] * iv + bias[o],
                    &out[((size_t)(b * CO + o)) * L_ + gn]);
            }
        }
    }
}

extern "C" void kernel_launch(void* const* d_in, const int* in_sizes, int n_in,
                              void* d_out, int out_size, void* d_ws, size_t ws_size,
                              hipStream_t stream) {
    const float* x    = (const float*)d_in[0];   // (8,128,112,112)
    const float* w    = (const float*)d_in[1];   // (256,128,3,3)
    const float* bias = (const float*)d_in[2];   // (256,)
    float* out = (float*)d_out;                  // (8,256,112,112)

    char* ws = (char*)d_ws;
    _Float16* wf   = (_Float16*)ws;                              // 589,824 B
    _Float16* x16t = (_Float16*)(ws + 589824);                   // 25,690,112 B
    float*    ss   = (float*)(ws + 589824 + 25690112);           // 401,408 B
    float*    inv  = (float*)(ws + 589824 + 25690112 + 401408);  // 401,408 B
    float*    zg   = (float*)(ws + 589824 + 25690112 + 2 * 401408); // 256 B zero guard

    cvt_weight_kernel<<<(CO * KD + 255) / 256, 256, 0, stream>>>(w, wf);
    {
        dim3 g(L_ / 32, C_ / 32, B_);            // (392, 4, 8)
        transpose_cvt_kernel<<<g, 256, 0, stream>>>(x, x16t);
    }
    chansq_kernel<<<(B_ * L_ + 255) / 256, 256, 0, stream>>>(x, ss);
    invnorm_kernel<<<(B_ * L_ + 255) / 256, 256, 0, stream>>>(ss, inv);
    zero_guard_kernel<<<1, 64, 0, stream>>>(zg);

    dim3 grid(L_ / NT, CO / MT, B_);             // (98, 2, 8)
    conv_wmma_kernel<<<grid, 256, 0, stream>>>(x16t, wf, inv, bias,
                                               (const _Float16*)zg, out);
}
